// VanillaRNN_60979945668922
// MI455X (gfx1250) — matrix operand order repaired and checked by reading, not compiled, not run on hardware
//
#include <hip/hip_runtime.h>
#include <hip/hip_bf16.h>

// VanillaRNN on MI455X (gfx1250): h_t = tanh(Whx x_t^T + Whh h_{t-1} + bh)
// Recurrent GEMM on v_wmma_f32_16x16x32_bf16 (fp32 accumulate), h kept
// transposed [B,H] in bf16. Tile staging via the CDNA5 Tensor Data Mover
// (tensor_load_to_lds + s_wait_tensorcnt), with per-lane async global->LDS
// DMA as fallback, 3-deep LDS pipeline either way.

#define B_ 2048
#define S_ 128
#define H_ 1024
#define C_ 10

typedef __attribute__((ext_vector_type(16))) __bf16 v16bf;
typedef __attribute__((ext_vector_type(8)))  float  v8f;

#if __has_builtin(__builtin_amdgcn_tensor_load_to_lds) && \
    __has_builtin(__builtin_amdgcn_s_wait_tensorcnt)
#define USE_TDM 1
#else
#define USE_TDM 0
#endif
#if __has_builtin(__builtin_amdgcn_global_load_async_to_lds_b128) && \
    __has_builtin(__builtin_amdgcn_s_wait_asynccnt)
#define USE_ASYNC 1
#else
#define USE_ASYNC 0
#endif

__device__ __forceinline__ unsigned short f2bf(float f) {
  unsigned int u = __float_as_uint(f);
  unsigned int r = (u + 0x7FFFu + ((u >> 16) & 1u)) >> 16;  // RNE
  return (unsigned short)r;
}
__device__ __forceinline__ float bf2f(unsigned short h) {
  return __uint_as_float(((unsigned int)h) << 16);
}

// ---------------------------------------------------------------- staging
#define LSTR 40  // LDS row stride in halfs (80B rows: conflict-free ds_load_b128)
#define BUFE (128 * LSTR)       // elements per LDS buffer
#define BUFB (BUFE * 2)         // bytes per LDS buffer

#if USE_TDM
typedef unsigned int tdm_v4u __attribute__((vector_size(16)));  // uint32x4
typedef int          tdm_v8i __attribute__((vector_size(32)));  // int32x8
typedef int          tdm_v4i __attribute__((vector_size(16)));  // int32x4

// One TDM descriptor load: 2D tile (32 bf16 wide x 128 rows) from a
// row-major tensor with 'stride_elems' elements per row, into LDS at
// 'lds_off', padding each 16-DWORD row with 4 DWORDs (=> LSTR stride).
__device__ __forceinline__ void tdm_load_tile(unsigned int lds_off, const void* gaddr,
                                              unsigned int stride_elems) {
  unsigned long long ga = (unsigned long long)gaddr;
  tdm_v4u g0;
  g0[0] = 1u;                                   // count=1, user descriptor
  g0[1] = lds_off;                              // lds_addr (bytes)
  g0[2] = (unsigned int)(ga & 0xffffffffu);     // global_addr[31:0]
  g0[3] = (unsigned int)((ga >> 32) & 0x1ffffffu) | (2u << 30);  // [56:32], type=2

  const unsigned long long td0 = 32, td1 = 128; // tensor dims = tile dims
  const unsigned long long st0 = stride_elems;  // dim0 stride (data_size units)
  unsigned long long q0 = (1ull << 16)          // data_size = 2 bytes
                        | (1ull << 20)          // pad_enable
                        | (3ull << 22)          // pad_interval: 16 DWORDs
                        | (3ull << 25)          // pad_amount:    4 DWORDs
                        | (td0 << 48);          // tensor_dim0[15:0]
  unsigned long long q1 = (td0 >> 16)           // tensor_dim0[31:16]
                        | (td1 << 16)           // tensor_dim1
                        | (32ull << 48);        // tile_dim0
  unsigned long long q2 = 128ull                // tile_dim1
                        | (0ull << 16)          // tile_dim2 unused
                        | (st0 << 32);          // tensor_dim0_stride[31:0]
  unsigned long long q3 = (st0 >> 32);          // stride[47:32]; dim1_stride=0
  tdm_v8i g1;
  g1[0] = (int)q0; g1[1] = (int)(q0 >> 32);
  g1[2] = (int)q1; g1[3] = (int)(q1 >> 32);
  g1[4] = (int)q2; g1[5] = (int)(q2 >> 32);
  g1[6] = (int)q3; g1[7] = (int)(q3 >> 32);
  tdm_v4i gz4 = {0, 0, 0, 0};                   // groups 2/3: 2D tensor
  tdm_v8i gz8 = {0, 0, 0, 0, 0, 0, 0, 0};       // extra group (6-arg form)
  __builtin_amdgcn_tensor_load_to_lds(g0, g1, gz4, gz4, gz8, 0);
}
#endif

#if USE_ASYNC
// Builtin prototype: (v4i AS1* src, v4i AS3* dst, imm offset, imm cpol)
typedef int v4i_gnu __attribute__((vector_size(16)));
typedef __attribute__((address_space(1))) v4i_gnu* gas_b128_t;
typedef __attribute__((address_space(3))) v4i_gnu* lds_b128_t;
__device__ __forceinline__ void async_cp16(const void* g, const void* l) {
  unsigned int lo = (unsigned int)(unsigned long long)l;  // low 32b = LDS offset
  __builtin_amdgcn_global_load_async_to_lds_b128(
      (gas_b128_t)(unsigned long long)g, (lds_b128_t)lo, 0, 0);
}
#endif

// ---------------------------------------------------------------- utilities
__global__ __launch_bounds__(256) void cvt_whh_bf16(const float* __restrict__ w,
                                                    unsigned short* __restrict__ o) {
  int i = blockIdx.x * 256 + threadIdx.x;
  if (i < H_ * H_) o[i] = f2bf(w[i]);
}

__global__ __launch_bounds__(256) void zero_h(unsigned short* __restrict__ h) {
  int i = blockIdx.x * 256 + threadIdx.x;
  if (i < B_ * H_) h[i] = 0;  // bf16 +0.0
}

// ---------------------------------------------------------------- RNN step
// Block tile: 128(M=h rows) x 128(N=batch). 8 waves: waveM 0..3 (32 rows),
// waveN 0..1 (64 cols). Wave = 2x4 WMMA tiles of 16x16. K chunked by 32.
#define KCH (H_ / 32)
#if USE_TDM || USE_ASYNC
#define NBUF 3
#else
#define NBUF 2
#endif

__global__ __launch_bounds__(256) void rnn_step(
    const unsigned short* __restrict__ Whh16,  // [H][H] bf16 row-major (M,K)
    const unsigned short* __restrict__ hPrev,  // [B][H] bf16 (N,K) transposed state
    unsigned short* __restrict__ hNext,        // [B][H] bf16
    const float* __restrict__ x,               // [B][S] fp32
    const float* __restrict__ Whx,             // [H] fp32
    const float* __restrict__ bh,              // [H] fp32
    int t) {
  __shared__ __align__(16) unsigned short As[NBUF][BUFE];  // Whh tile  [128m][32k]
  __shared__ __align__(16) unsigned short Bs[NBUF][BUFE];  // hPrev tile[128b][32k]

  const int tid   = threadIdx.x;
  const int lane  = tid & 31;
  const int wave  = tid >> 5;
  const int l15   = lane & 15;
  const int lhi   = lane >> 4;       // 0|1 : half-wave select
  const int waveM = wave & 3;
  const int waveN = wave >> 2;
  const int mBase = blockIdx.y * 128;
  const int bBase = blockIdx.x * 128;

  const unsigned short* Ag = Whh16 + (size_t)mBase * H_;
  const unsigned short* Bg = hPrev + (size_t)bBase * H_;

  v8f acc[2][4];
#pragma unroll
  for (int mi = 0; mi < 2; ++mi)
#pragma unroll
    for (int ni = 0; ni < 4; ++ni) acc[mi][ni] = (v8f)0.0f;

  union Frag { v16bf v; uint4 q[2]; };
  // Per-wave frag read bases (element offsets inside one buffer)
  const int aRow0 = (waveM * 32 + l15) * LSTR;          // mi stride = 16*LSTR
  const int bRow0 = (waveN * 64 + l15) * LSTR;          // ni stride = 16*LSTR

#if USE_TDM
  // ---- Tensor Data Mover pipeline: wave0 issues 2 TDM tile loads/chunk ----
  const unsigned asBase = (unsigned int)(unsigned long long)(const void*)&As[0][0];
  const unsigned bsBase = (unsigned int)(unsigned long long)(const void*)&Bs[0][0];
  if (wave == 0) {
    tdm_load_tile(asBase + 0 * BUFB, Ag + 0 * 32, H_);
    tdm_load_tile(bsBase + 0 * BUFB, Bg + 0 * 32, H_);
    tdm_load_tile(asBase + 1 * BUFB, Ag + 1 * 32, H_);
    tdm_load_tile(bsBase + 1 * BUFB, Bg + 1 * 32, H_);
  }
  int bufR = 0;   // buffer holding chunk kc
  int bufW = 2;   // buffer to receive chunk kc+2
#pragma unroll 1
  for (int kc = 0; kc < KCH; ++kc) {
    if (wave == 0) {
      // TDM completes in-order per wave: <=2 outstanding => chunk kc landed.
      if (kc + 1 < KCH) __builtin_amdgcn_s_wait_tensorcnt(2);
      else              __builtin_amdgcn_s_wait_tensorcnt(0);
    }
    __syncthreads();                       // chunk kc visible to all waves
    if (wave == 0 && kc + 2 < KCH) {       // its buffer's readers are done
      const int ko = (kc + 2) * 32;
      tdm_load_tile(asBase + bufW * BUFB, Ag + ko, H_);
      tdm_load_tile(bsBase + bufW * BUFB, Bg + ko, H_);
    }
    const unsigned short* aP = &As[0][0] + bufR * BUFE;
    const unsigned short* bP = &Bs[0][0] + bufR * BUFE;

    Frag aF[2];
#pragma unroll
    for (int mi = 0; mi < 2; ++mi) {
      const unsigned short* rp = aP + aRow0 + mi * 16 * LSTR;
      aF[mi].q[0] = *(const uint4*)(rp + lhi * 8);
      aF[mi].q[1] = *(const uint4*)(rp + 16 + lhi * 8);
    }
    Frag bF[4];
#pragma unroll
    for (int ni = 0; ni < 4; ++ni) {
      const unsigned short* rp = bP + bRow0 + ni * 16 * LSTR;
      bF[ni].q[0] = *(const uint4*)(rp + lhi * 8);
      bF[ni].q[1] = *(const uint4*)(rp + 16 + lhi * 8);
    }
#pragma unroll
    for (int mi = 0; mi < 2; ++mi)
#pragma unroll
      for (int ni = 0; ni < 4; ++ni)
        acc[mi][ni] = __builtin_amdgcn_wmma_f32_16x16x32_bf16(
            false, aF[mi].v, false, bF[ni].v, (short)0, acc[mi][ni], false, false);
    bufR = (bufR == 2) ? 0 : bufR + 1;
    bufW = (bufW == 2) ? 0 : bufW + 1;
  }
#elif USE_ASYNC
  // ---- per-lane async DMA pipeline (ASYNCcnt) ----
  const int r0 = tid >> 2;
  const int r1 = r0 + 64;
  const int c0 = (tid & 3) * 8;
  auto issue_chunk = [&](int kc, int buf) {
    const int ko = kc * 32;
    async_cp16(Ag + r0 * H_ + ko + c0, &As[buf][r0 * LSTR + c0]);
    async_cp16(Ag + r1 * H_ + ko + c0, &As[buf][r1 * LSTR + c0]);
    async_cp16(Bg + r0 * H_ + ko + c0, &Bs[buf][r0 * LSTR + c0]);
    async_cp16(Bg + r1 * H_ + ko + c0, &Bs[buf][r1 * LSTR + c0]);
  };
  issue_chunk(0, 0);
  issue_chunk(1, 1);
  int bufR = 0, bufW = 2;
#pragma unroll 1
  for (int kc = 0; kc < KCH; ++kc) {
    if (kc + 1 < KCH) __builtin_amdgcn_s_wait_asynccnt(4);
    else              __builtin_amdgcn_s_wait_asynccnt(0);
    __syncthreads();
    if (kc + 2 < KCH) issue_chunk(kc + 2, bufW);
    const unsigned short* aP = &As[0][0] + bufR * BUFE;
    const unsigned short* bP = &Bs[0][0] + bufR * BUFE;
    Frag aF[2];
#pragma unroll
    for (int mi = 0; mi < 2; ++mi) {
      const unsigned short* rp = aP + aRow0 + mi * 16 * LSTR;
      aF[mi].q[0] = *(const uint4*)(rp + lhi * 8);
      aF[mi].q[1] = *(const uint4*)(rp + 16 + lhi * 8);
    }
    Frag bF[4];
#pragma unroll
    for (int ni = 0; ni < 4; ++ni) {
      const unsigned short* rp = bP + bRow0 + ni * 16 * LSTR;
      bF[ni].q[0] = *(const uint4*)(rp + lhi * 8);
      bF[ni].q[1] = *(const uint4*)(rp + 16 + lhi * 8);
    }
#pragma unroll
    for (int mi = 0; mi < 2; ++mi)
#pragma unroll
      for (int ni = 0; ni < 4; ++ni)
        acc[mi][ni] = __builtin_amdgcn_wmma_f32_16x16x32_bf16(
            false, aF[mi].v, false, bF[ni].v, (short)0, acc[mi][ni], false, false);
    bufR = (bufR == 2) ? 0 : bufR + 1;
    bufW = (bufW == 2) ? 0 : bufW + 1;
  }
#else
  // ---- fallback: synchronous global->VGPR->LDS double buffering ----
  const int r0 = tid >> 2;
  const int r1 = r0 + 64;
  const int c0 = (tid & 3) * 8;
  uint4 aR0 = *(const uint4*)(Ag + r0 * H_ + c0);
  uint4 aR1 = *(const uint4*)(Ag + r1 * H_ + c0);
  uint4 bR0 = *(const uint4*)(Bg + r0 * H_ + c0);
  uint4 bR1 = *(const uint4*)(Bg + r1 * H_ + c0);
#pragma unroll 1
  for (int kc = 0; kc < KCH; ++kc) {
    const int buf = kc & 1;
    *(uint4*)(&As[buf][r0 * LSTR + c0]) = aR0;
    *(uint4*)(&As[buf][r1 * LSTR + c0]) = aR1;
    *(uint4*)(&Bs[buf][r0 * LSTR + c0]) = bR0;
    *(uint4*)(&Bs[buf][r1 * LSTR + c0]) = bR1;
    if (kc + 1 < KCH) {
      const int ko = (kc + 1) * 32;
      aR0 = *(const uint4*)(Ag + r0 * H_ + ko + c0);
      aR1 = *(const uint4*)(Ag + r1 * H_ + ko + c0);
      bR0 = *(const uint4*)(Bg + r0 * H_ + ko + c0);
      bR1 = *(const uint4*)(Bg + r1 * H_ + ko + c0);
    }
    __syncthreads();
    const unsigned short* aP = &As[buf][0];
    const unsigned short* bP = &Bs[buf][0];
    Frag aF[2];
#pragma unroll
    for (int mi = 0; mi < 2; ++mi) {
      const unsigned short* rp = aP + aRow0 + mi * 16 * LSTR;
      aF[mi].q[0] = *(const uint4*)(rp + lhi * 8);
      aF[mi].q[1] = *(const uint4*)(rp + 16 + lhi * 8);
    }
    Frag bF[4];
#pragma unroll
    for (int ni = 0; ni < 4; ++ni) {
      const unsigned short* rp = bP + bRow0 + ni * 16 * LSTR;
      bF[ni].q[0] = *(const uint4*)(rp + lhi * 8);
      bF[ni].q[1] = *(const uint4*)(rp + 16 + lhi * 8);
    }
#pragma unroll
    for (int mi = 0; mi < 2; ++mi)
#pragma unroll
      for (int ni = 0; ni < 4; ++ni)
        acc[mi][ni] = __builtin_amdgcn_wmma_f32_16x16x32_bf16(
            false, aF[mi].v, false, bF[ni].v, (short)0, acc[mi][ni], false, false);
    __syncthreads();
  }
#endif

  // Epilogue: + Whx[m]*x[b,t] + bh[m], tanh, ->bf16, 128-bit stores.
  // D layout: lane holds N=l15; VGPR r holds M = r + 8*lhi => 8 consecutive m.
#pragma unroll
  for (int mi = 0; mi < 2; ++mi) {
    const int mstart = mBase + waveM * 32 + mi * 16 + lhi * 8;
    float wx[8], bb[8];
#pragma unroll
    for (int r = 0; r < 8; ++r) { wx[r] = Whx[mstart + r]; bb[r] = bh[mstart + r]; }
#pragma unroll
    for (int ni = 0; ni < 4; ++ni) {
      const int b  = bBase + waveN * 64 + ni * 16 + l15;
      const float xv = x[(size_t)b * S_ + t];
      unsigned int q[4];
#pragma unroll
      for (int r = 0; r < 8; r += 2) {
        float v0 = tanhf(acc[mi][ni][r]     + wx[r]     * xv + bb[r]);
        float v1 = tanhf(acc[mi][ni][r + 1] + wx[r + 1] * xv + bb[r + 1]);
        q[r >> 1] = (unsigned int)f2bf(v0) | ((unsigned int)f2bf(v1) << 16);
      }
      *(uint4*)(hNext + (size_t)b * H_ + mstart) = make_uint4(q[0], q[1], q[2], q[3]);
    }
  }
}

// ---------------------------------------------------------------- head
__global__ __launch_bounds__(256) void logits_kernel(
    const unsigned short* __restrict__ hT,  // [B][H] bf16
    const float* __restrict__ Wph,          // [C][H]
    const float* __restrict__ bp,           // [C]
    float* __restrict__ out) {              // [B][C]
  int idx = blockIdx.x * 256 + threadIdx.x;
  if (idx >= B_ * C_) return;
  int b = idx / C_, c = idx % C_;
  const unsigned short* hp = hT + (size_t)b * H_;
  const float* wp = Wph + (size_t)c * H_;
  float s = 0.0f;
#pragma unroll 8
  for (int k = 0; k < H_; ++k) s += bf2f(hp[k]) * wp[k];
  out[idx] = s + bp[c];
}

// ---------------------------------------------------------------- launcher
extern "C" void kernel_launch(void* const* d_in, const int* in_sizes, int n_in,
                              void* d_out, int out_size, void* d_ws, size_t ws_size,
                              hipStream_t stream) {
  const float* x   = (const float*)d_in[0];
  const float* Whx = (const float*)d_in[1];
  const float* Whh = (const float*)d_in[2];
  const float* bh  = (const float*)d_in[3];
  const float* Wph = (const float*)d_in[4];
  const float* bp  = (const float*)d_in[5];
  float* out = (float*)d_out;

  char* ws = (char*)d_ws;
  unsigned short* Whh16 = (unsigned short*)ws;                       // 2 MB
  unsigned short* h0    = (unsigned short*)(ws + 2u * 1024 * 1024);  // 4 MB
  unsigned short* h1    = (unsigned short*)(ws + 6u * 1024 * 1024);  // 4 MB

  cvt_whh_bf16<<<(H_ * H_) / 256, 256, 0, stream>>>(Whh, Whh16);
  zero_h<<<(B_ * H_) / 256, 256, 0, stream>>>(h0);

  dim3 grid(B_ / 128, H_ / 128);  // 16 x 8 = 128 blocks of 8 waves
  for (int t = 0; t < S_; ++t) {
    unsigned short* hp = (t & 1) ? h1 : h0;
    unsigned short* hn = (t & 1) ? h0 : h1;
    rnn_step<<<grid, 256, 0, stream>>>(Whh16, hp, hn, x, Whx, bh, t);
  }
  // S_ even -> final state in h0
  logits_kernel<<<(B_ * C_ + 255) / 256, 256, 0, stream>>>(h0, Wph, bp, out);
}
